// StackedLSTM_18150531793481
// MI455X (gfx1250) — compile-verified, matrix operand order checked by reading, not run on hardware
//
#include <hip/hip_runtime.h>

// Problem constants (match the reference).
#define B_  128
#define T_  128
#define DI_ 1024
#define H_  1024
#define L_  4
#define KK  1024   // contraction depth of every GEMM here (DI_ == H_ == 1024)

typedef __attribute__((ext_vector_type(16))) __bf16 v16bf;
typedef __attribute__((ext_vector_type(8)))  __bf16 bf16x8;
typedef __attribute__((ext_vector_type(8)))  float  v8f;

union FragBF { bf16x8 h[2]; v16bf v; };

__device__ __forceinline__ float sigf(float x) { return 1.0f / (1.0f + __expf(-x)); }

// LDS staging for the shared weight tile: 64 rows (N) x 32 K bf16 = 64 B/row,
// padded to 80 B/row (16B-aligned chunks, 20-bank stride -> conflict-free).
#define LDSROW    80
#define TILEBYTES (64 * LDSROW)

// ---------------------------------------------------------------------------
// C(MxN) = A(MxKK) @ Bw(NxKK)^T  [+ C | + bias],  bf16 operands, f32 accum.
// Block = 4 waves = one 64(M) x 64(N) strip, one ng per block.
// Weight tiles are async-copied global->LDS (ASYNCcnt, FIFO), double buffered;
// A fragments live in named registers (no runtime-indexed arrays!).
// ---------------------------------------------------------------------------
__global__ __launch_bounds__(128)
void wmma_gemm_bf16(const __bf16* __restrict__ A, long lda,
                    const __bf16* __restrict__ Bw, long ldb,
                    const float*  __restrict__ bias,
                    float* __restrict__ C, long ldc,
                    int Mtiles, int Ngroups, int accum)
{
    __shared__ __align__(16) char ldsB[2 * TILEBYTES];

    const int tid  = threadIdx.x;
    const int wave = tid >> 5;
    const int lane = tid & 31;
    const int mg4  = Mtiles >> 2;              // blocks per ng
    const int ng   = blockIdx.x / mg4;         // consecutive blocks sweep M: weight L2 reuse
    const int mg   = blockIdx.x - ng * mg4;
    const int mt   = mg * 4 + wave;            // this wave's 16-row m-tile
    const int half = lane >> 4;
    const int l16  = lane & 31 & 15;

    // ---- async global->LDS staging: thread stages 16B chunk `sub` of weight
    //      rows bn and bn+32 (256 chunks per 4KB tile, 2 per thread).
    const int bn  = tid >> 2;
    const int sub = tid & 3;
    const __bf16* gB0 = Bw + (size_t)(ng * 64 + bn)      * ldb + sub * 8;
    const __bf16* gB1 = Bw + (size_t)(ng * 64 + bn + 32) * ldb + sub * 8;
    const unsigned ldsbase = (unsigned)(size_t)(void*)ldsB;  // low 32 bits = LDS byte offset
    const unsigned lB0 = ldsbase + bn * LDSROW + sub * 16;
    const unsigned lB1 = ldsbase + (bn + 32) * LDSROW + sub * 16;

    auto issue_tile = [&](int k, int buf) {
        const unsigned d0 = lB0 + buf * TILEBYTES;
        const unsigned d1 = lB1 + buf * TILEBYTES;
        const __bf16* s0 = gB0 + k;
        const __bf16* s1 = gB1 + k;
        asm volatile("global_load_async_to_lds_b128 %0, %1, off\n\t"
                     "global_load_async_to_lds_b128 %2, %3, off"
                     :: "v"(d0), "v"(s0), "v"(d1), "v"(s1)
                     : "memory");
    };

    // ---- accumulator init: C (accumulate), bias, or zero.
    v8f acc[4];
    #pragma unroll
    for (int nt = 0; nt < 4; ++nt) {
        const int ncol = ng * 64 + nt * 16 + l16;
        if (accum) {
            #pragma unroll
            for (int v = 0; v < 8; ++v)
                acc[nt][v] = C[(size_t)(mt * 16 + half * 8 + v) * ldc + ncol];
        } else {
            const float bv = bias ? bias[ncol] : 0.0f;
            #pragma unroll
            for (int v = 0; v < 8; ++v) acc[nt][v] = bv;
        }
    }

    // B-fragment reads from LDS + 4 WMMAs sharing one A fragment.
    auto compute = [&](const FragBF& af, const char* bufp) {
        #pragma unroll
        for (int nt = 0; nt < 4; ++nt) {
            FragBF b;
            const char* row = bufp + (nt * 16 + l16) * LDSROW + half * 16;
            b.h[0] = *(const bf16x8*)(row);
            b.h[1] = *(const bf16x8*)(row + 32);
            acc[nt] = __builtin_amdgcn_wmma_f32_16x16x32_bf16(
                false, af.v, false, b.v, (short)0, acc[nt], false, false);
        }
    };

    // ---- A fragment base: per lane two contiguous 16B runs (CDNA5 16-bit layout).
    const __bf16* pa = A + (size_t)(mt * 16 + l16) * lda + half * 8;
    const char* buf0 = ldsB;
    const char* buf1 = ldsB + TILEBYTES;

    // ---- software-pipelined main loop: two tiles in flight, named registers.
    issue_tile(0, 0);
    issue_tile(32, 1);
    FragBF a0, a1;
    a0.h[0] = *(const bf16x8*)(pa);
    a0.h[1] = *(const bf16x8*)(pa + 16);
    a1.h[0] = *(const bf16x8*)(pa + 32);
    a1.h[1] = *(const bf16x8*)(pa + 48);

    for (int k = 0; k + 64 < KK; k += 64) {
        // even tile k (buf0): our oldest 2 async ops are buf0's pair.
        asm volatile("s_wait_asynccnt 0x2" ::: "memory");
        __syncthreads();
        compute(a0, buf0);
        __syncthreads();                 // all waves done with buf0 before refill
        issue_tile(k + 64, 0);
        a0.h[0] = *(const bf16x8*)(pa + k + 64);
        a0.h[1] = *(const bf16x8*)(pa + k + 80);

        // odd tile k+32 (buf1)
        asm volatile("s_wait_asynccnt 0x2" ::: "memory");
        __syncthreads();
        compute(a1, buf1);
        __syncthreads();
        issue_tile(k + 96, 1);
        a1.h[0] = *(const bf16x8*)(pa + k + 96);
        a1.h[1] = *(const bf16x8*)(pa + k + 112);
    }
    // epilogue: tiles KK-64 (buf0) and KK-32 (buf1)
    asm volatile("s_wait_asynccnt 0x2" ::: "memory");
    __syncthreads();
    compute(a0, buf0);
    __syncthreads();
    asm volatile("s_wait_asynccnt 0x0" ::: "memory");
    __syncthreads();
    compute(a1, buf1);

    #pragma unroll
    for (int nt = 0; nt < 4; ++nt) {
        const int ncol = ng * 64 + nt * 16 + l16;
        #pragma unroll
        for (int v = 0; v < 8; ++v)
            C[(size_t)(mt * 16 + half * 8 + v) * ldc + ncol] = acc[nt][v];
    }
}

// ---------------------------------------------------------------------------
__global__ void f32_to_bf16_kernel(const float* __restrict__ in,
                                   __bf16* __restrict__ out, size_t n)
{
    size_t i = (size_t)blockIdx.x * blockDim.x + threadIdx.x;
    const size_t stride = (size_t)gridDim.x * blockDim.x;
    for (; i < n; i += stride) out[i] = (__bf16)in[i];
}

// Layer-0 plain LSTM gates. proj rows (b*T+t) of width 4H hold i,f,g,o preacts.
__global__ void lstm_gate0(const float* __restrict__ proj, int t,
                           float* __restrict__ hs, float* __restrict__ cs,
                           __bf16* __restrict__ hsb)
{
    const int idx = blockIdx.x * blockDim.x + threadIdx.x;
    const int b = idx / H_;
    const int j = idx - b * H_;
    const float* g = proj + (size_t)(b * T_ + t) * (4 * H_);
    const float gi = g[j], gf = g[H_ + j], gg = g[2 * H_ + j], go = g[3 * H_ + j];
    const float cp = (t > 0) ? cs[(size_t)(b * T_ + t - 1) * H_ + j] : 0.0f;
    const float c = sigf(gf) * cp + sigf(gi) * tanhf(gg);
    const float h = sigf(go) * tanhf(c);
    const size_t o = (size_t)(b * T_ + t) * H_ + j;
    hs[o] = h; cs[o] = c; hsb[o] = (__bf16)h;
}

// CALSTM gates. proj rows of width 5H hold i, f_prev, f_low, u, o preacts.
__global__ void calstm_gate(const float* __restrict__ proj, int t,
                            const float* __restrict__ cs_low,
                            float* __restrict__ hs, float* __restrict__ cs,
                            __bf16* __restrict__ hsb)
{
    const int idx = blockIdx.x * blockDim.x + threadIdx.x;
    const int b = idx / H_;
    const int j = idx - b * H_;
    const float* g = proj + (size_t)(b * T_ + t) * (5 * H_);
    const float gi = g[j], gfp = g[H_ + j], gfl = g[2 * H_ + j],
                gu = g[3 * H_ + j], go = g[4 * H_ + j];
    const float cp = (t > 0) ? cs[(size_t)(b * T_ + t - 1) * H_ + j] : 0.0f;
    const float cl = cs_low[(size_t)(b * T_ + t) * H_ + j];
    const float c = cp * sigf(gfp + 1.0f) + cl * sigf(gfl + 1.0f) + tanhf(gu) * sigf(gi);
    const float h = sigf(go) * tanhf(c);
    const size_t o = (size_t)(b * T_ + t) * H_ + j;
    hs[o] = h; cs[o] = c; hsb[o] = (__bf16)h;
}

__global__ void gather_final(const int* __restrict__ len,
                             const float* __restrict__ hs, const float* __restrict__ cs,
                             float* __restrict__ oh, float* __restrict__ oc)
{
    const int idx = blockIdx.x * blockDim.x + threadIdx.x;
    const int b = idx / H_;
    const int j = idx - b * H_;
    const int t = len[b] - 1;
    oh[(size_t)b * H_ + j] = hs[(size_t)(b * T_ + t) * H_ + j];
    oc[(size_t)b * H_ + j] = cs[(size_t)(b * T_ + t) * H_ + j];
}

// ---------------------------------------------------------------------------
extern "C" void kernel_launch(void* const* d_in, const int* in_sizes, int n_in,
                              void* d_out, int out_size, void* d_ws, size_t ws_size,
                              hipStream_t stream)
{
    (void)in_sizes; (void)n_in; (void)out_size; (void)ws_size;

    const float* x   = (const float*)d_in[0];
    const int*   len = (const int*)  d_in[1];
    const float* Wih = (const float*)d_in[2];
    const float* Whh = (const float*)d_in[3];
    const float* b0  = (const float*)d_in[4];
    const float* Wca = (const float*)d_in[5];
    const float* bca = (const float*)d_in[6];

    // Workspace carve-out (~784 MB total), 256B aligned slices.
    char* ws = (char*)d_ws;
    size_t off = 0;
    auto take = [&](size_t bytes) -> char* {
        char* p = ws + off;
        off = (off + bytes + 255) & ~(size_t)255;
        return p;
    };
    __bf16* xb   = (__bf16*)take((size_t)B_ * T_ * DI_ * sizeof(__bf16));
    __bf16* wihb = (__bf16*)take((size_t)4 * H_ * DI_ * sizeof(__bf16));
    __bf16* whhb = (__bf16*)take((size_t)4 * H_ * H_ * sizeof(__bf16));
    __bf16* wcab = (__bf16*)take((size_t)(L_ - 1) * 5 * H_ * 2 * H_ * sizeof(__bf16));
    float*  proj = (float*) take((size_t)B_ * T_ * 5 * H_ * sizeof(float));
    float*  hsA  = (float*) take((size_t)B_ * T_ * H_ * sizeof(float));
    float*  csA  = (float*) take((size_t)B_ * T_ * H_ * sizeof(float));
    float*  hsB  = (float*) take((size_t)B_ * T_ * H_ * sizeof(float));
    float*  csB  = (float*) take((size_t)B_ * T_ * H_ * sizeof(float));
    __bf16* hsbA = (__bf16*)take((size_t)B_ * T_ * H_ * sizeof(__bf16));
    __bf16* hsbB = (__bf16*)take((size_t)B_ * T_ * H_ * sizeof(__bf16));

    float* out_hs = (float*)d_out;
    float* out_cs = out_hs + (size_t)B_ * T_ * H_;
    float* out_hf = out_cs + (size_t)B_ * T_ * H_;
    float* out_cf = out_hf + (size_t)L_ * B_ * H_;

    // Precision conversions (once per call).
    f32_to_bf16_kernel<<<4096, 256, 0, stream>>>(x,   xb,   (size_t)B_ * T_ * DI_);
    f32_to_bf16_kernel<<<2048, 256, 0, stream>>>(Wih, wihb, (size_t)4 * H_ * DI_);
    f32_to_bf16_kernel<<<2048, 256, 0, stream>>>(Whh, whhb, (size_t)4 * H_ * H_);
    f32_to_bf16_kernel<<<4096, 256, 0, stream>>>(Wca, wcab, (size_t)(L_ - 1) * 5 * H_ * 2 * H_);

    // GEMM grid: one block per (4 m-tiles) x (64-col n-group); exact fit.
    auto gemm_blocks = [](int Mtiles, int Ngroups) { return (Mtiles >> 2) * Ngroups; };

    // ---- Layer 0: big input projection (parallel over all t), then recurrence.
    {
        const int Mt = (B_ * T_) / 16, Ng = (4 * H_) / 64;
        wmma_gemm_bf16<<<gemm_blocks(Mt, Ng), 128, 0, stream>>>(
            xb, DI_, wihb, DI_, b0, proj, 4 * H_, Mt, Ng, 0);
    }
    for (int t = 0; t < T_; ++t) {
        if (t > 0) {
            const int Mt = B_ / 16, Ng = (4 * H_) / 64;
            wmma_gemm_bf16<<<gemm_blocks(Mt, Ng), 128, 0, stream>>>(
                hsbA + (size_t)(t - 1) * H_, (long)T_ * H_,
                whhb, H_, nullptr,
                proj + (size_t)t * 4 * H_, (long)T_ * 4 * H_,
                Mt, Ng, 1);
        }
        lstm_gate0<<<(B_ * H_) / 256, 256, 0, stream>>>(proj, t, hsA, csA, hsbA);
    }
    gather_final<<<(B_ * H_) / 256, 256, 0, stream>>>(len, hsA, csA, out_hf, out_cf);

    // ---- Layers 1..3: CALSTM. Lower-h projection is batched; only the
    //      h_prev half of the gate GEMM sits on the sequential critical path.
    float*  hs_prev  = hsA;  float* cs_prev = csA;  __bf16* hsb_prev = hsbA;
    for (int l = 1; l < L_; ++l) {
        float* hs_cur;  float* cs_cur;  __bf16* hsb_cur;
        if (l & 1) { hs_cur = hsB; cs_cur = csB; hsb_cur = hsbB; }
        else       { hs_cur = hsA; cs_cur = csA; hsb_cur = hsbA; }
        if (l == L_ - 1) { hs_cur = out_hs; cs_cur = out_cs; }  // last layer: straight to d_out

        const __bf16* wl = wcab + (size_t)(l - 1) * 5 * H_ * 2 * H_;
        const float*  bl = bca  + (size_t)(l - 1) * 5 * H_;

        {   // batched: hs_low @ W[:, :H]^T + b   (M=B*T, N=5H, K=H)
            const int Mt = (B_ * T_) / 16, Ng = (5 * H_) / 64;
            wmma_gemm_bf16<<<gemm_blocks(Mt, Ng), 128, 0, stream>>>(
                hsb_prev, H_, wl, 2 * H_, bl, proj, 5 * H_, Mt, Ng, 0);
        }
        for (int t = 0; t < T_; ++t) {
            if (t > 0) {  // critical path: h_prev @ W[:, H:]^T accumulated into proj
                const int Mt = B_ / 16, Ng = (5 * H_) / 64;
                wmma_gemm_bf16<<<gemm_blocks(Mt, Ng), 128, 0, stream>>>(
                    hsb_cur + (size_t)(t - 1) * H_, (long)T_ * H_,
                    wl + H_, 2 * H_, nullptr,
                    proj + (size_t)t * 5 * H_, (long)T_ * 5 * H_,
                    Mt, Ng, 1);
            }
            calstm_gate<<<(B_ * H_) / 256, 256, 0, stream>>>(
                proj, t, cs_prev, hs_cur, cs_cur, hsb_cur);
        }
        gather_final<<<(B_ * H_) / 256, 256, 0, stream>>>(
            len, hs_cur, cs_cur, out_hf + (size_t)l * B_ * H_, out_cf + (size_t)l * B_ * H_);

        hs_prev = hs_cur; cs_prev = cs_cur; hsb_prev = hsb_cur;
    }
}
